// MicrotubuleAttention_15401752724150
// MI455X (gfx1250) — matrix operand order repaired and checked by reading, not compiled, not run on hardware
//
#include <hip/hip_runtime.h>
#include <hip/hip_bf16.h>

typedef __attribute__((ext_vector_type(16))) __bf16 v16bf;
typedef __attribute__((ext_vector_type(8)))  float  v8f;
typedef __attribute__((ext_vector_type(4)))  int    v4i;
typedef unsigned short u16;
typedef unsigned int   u32;

#define D_MODEL 2048
#define N_HEADS 16
#define N_KV_HEADS 4
#define D_HEAD 128
#define SEQ_T 2048
#define RANK 32
#define QKV_N (N_HEADS*D_HEAD + 2*N_KV_HEADS*D_HEAD)   // 3072

// ---------------- CDNA5 async global->LDS path (guarded) ----------------
#if defined(__AMDGCN__) && __has_builtin(__builtin_amdgcn_global_load_async_to_lds_b128)
#define ASYNC_LDS 1
#else
#define ASYNC_LDS 0
#endif

__device__ __forceinline__ void stage16(u16* dst_lds, const u16* src_global) {
#if ASYNC_LDS
    __builtin_amdgcn_global_load_async_to_lds_b128(
        (__attribute__((address_space(1))) v4i*)(src_global),
        (__attribute__((address_space(3))) v4i*)(dst_lds), 0, 0);
#else
    *(uint4*)dst_lds = *(const uint4*)src_global;
#endif
}
__device__ __forceinline__ void stage_fence() {
#if ASYNC_LDS
#if __has_builtin(__builtin_amdgcn_s_wait_asynccnt)
    __builtin_amdgcn_s_wait_asynccnt(0);
#else
    asm volatile("s_wait_asynccnt 0x0" ::: "memory");
#endif
#endif
}

// ---------------- bf16 helpers ----------------
__device__ __forceinline__ u16 f2bf(float f) {
    u32 u = __float_as_uint(f);
    u32 r = u + 0x7fffu + ((u >> 16) & 1u);
    return (u16)(r >> 16);
}
__device__ __forceinline__ float bf2f(u16 h) {
    return __uint_as_float(((u32)h) << 16);
}

// ---------------- WMMA fragment loads (wave32 layouts, ISA 7.12.2) -------
// A 16x32 bf16: lane m=lane&15, half=lane>>4; K runs [8h..8h+7] and [16+8h..16+8h+7]
__device__ __forceinline__ v16bf ld_frag_a(const u16* base, int stride) {
    int lane = threadIdx.x & 31;
    int m = lane & 15, h = lane >> 4;
    union { v16bf v; uint4 q[2]; } f;
    const u16* p = base + m * stride + 8 * h;
    f.q[0] = *(const uint4*)(p);
    f.q[1] = *(const uint4*)(p + 16);
    return f.v;
}
// B 32x16 bf16 given Bt[n][k] in LDS: lane n=lane&15, half=lane>>4; K runs [8h..8h+7] and [16+8h..16+8h+7]
__device__ __forceinline__ v16bf ld_frag_b(const u16* base, int stride) {
    int lane = threadIdx.x & 31;
    int n = lane & 15, h = lane >> 4;
    union { v16bf v; uint4 q[2]; } f;
    const u16* p = base + n * stride + 8 * h;
    f.q[0] = *(const uint4*)(p);
    f.q[1] = *(const uint4*)(p + 16);
    return f.v;
}
__device__ __forceinline__ v8f wmma_bf16(v16bf a, v16bf b, v8f c) {
    return __builtin_amdgcn_wmma_f32_16x16x32_bf16(false, a, false, b, (short)0, c, false, false);
}
__device__ __forceinline__ v8f v8f_zero() {
    v8f s;
#pragma unroll
    for (int r = 0; r < 8; ++r) s[r] = 0.0f;
    return s;
}

// ---------------- elementwise conversion kernels ----------------
__global__ void cvt_bf16_kernel(const float* __restrict__ src, u16* __restrict__ dst, int n) {
    int i = blockIdx.x * blockDim.x + threadIdx.x;
    if (i < n) dst[i] = f2bf(src[i]);
}
// src[K][N] fp32 -> dst[N][K] bf16
__global__ void transpose_cvt_kernel(const float* __restrict__ src, u16* __restrict__ dst, int K, int N) {
    int i = blockIdx.x * blockDim.x + threadIdx.x;
    if (i < K * N) {
        int k = i / N, n = i - k * N;
        dst[(size_t)n * K + k] = f2bf(src[i]);
    }
}

// ---------------- rank-32 bias factors: A = x@WA, B = x@WB ----------------
__global__ void pol_ab_kernel(const float* __restrict__ x, const float* __restrict__ WA,
                              const float* __restrict__ WB, u16* __restrict__ Abf,
                              u16* __restrict__ Bbf) {
    int idx = blockIdx.x * blockDim.x + threadIdx.x;       // T*RANK
    if (idx >= SEQ_T * RANK) return;
    int t = idx >> 5, r = idx & 31;
    const float* xr = x + (size_t)t * D_MODEL;
    float sa = 0.f, sb = 0.f;
    for (int k = 0; k < D_MODEL; ++k) {
        float xv = xr[k];
        sa = fmaf(xv, WA[(size_t)k * RANK + r], sa);
        sb = fmaf(xv, WB[(size_t)k * RANK + r], sb);
    }
    Abf[(size_t)t * RANK + r] = f2bf(sa);
    Bbf[(size_t)t * RANK + r] = f2bf(sb);
}

// ---------------- generic TN WMMA GEMM: C[M,N] f32 = A[M,K] bf16 @ Bt[N,K] bf16 ----
// block tile 128x128, 8 waves (4x2), wave tile 32x64, double-buffered LDS
__global__ __launch_bounds__(256) void gemm_tn_bf16(const u16* __restrict__ A,
                                                    const u16* __restrict__ Bt,
                                                    float* __restrict__ C,
                                                    int M, int N, int K) {
    extern __shared__ char smem[];
    u16* sA = (u16*)smem;                       // [2][128][32]
    u16* sB = (u16*)(smem + 2 * 128 * 32 * 2);  // [2][128][32]
    const int TILE = 128 * 32;

    int tid = threadIdx.x;
    int wave = tid >> 5;
    int wr = wave >> 1, wc = wave & 1;
    int bm = blockIdx.y * 128, bn = blockIdx.x * 128;

    v8f acc[2][4];
#pragma unroll
    for (int i = 0; i < 2; ++i)
#pragma unroll
        for (int j = 0; j < 4; ++j) acc[i][j] = v8f_zero();

    auto loadTile = [&](u16* dst, const u16* src, int ld) {
#pragma unroll
        for (int c = 0; c < 2; ++c) {
            int idx = tid + c * 256;        // 512 x (8 u16) chunks
            int r = idx >> 2;
            int col = (idx & 3) * 8;
            stage16(dst + r * 32 + col, src + (size_t)r * ld + col);
        }
    };

    int nk = K / 32;
    int buf = 0;
    loadTile(sA, A + (size_t)bm * K, K);
    loadTile(sB, Bt + (size_t)bn * K, K);
    stage_fence();
    __syncthreads();

    for (int kt = 0; kt < nk; ++kt) {
        if (kt + 1 < nk) {
            loadTile(sA + (buf ^ 1) * TILE, A + (size_t)bm * K + (kt + 1) * 32, K);
            loadTile(sB + (buf ^ 1) * TILE, Bt + (size_t)bn * K + (kt + 1) * 32, K);
        }
        if (kt + 2 < nk) {  // L2 prefetch of the tile after next (global_prefetch_b8)
            int r = tid >> 2, col = (tid & 3) * 8;
            __builtin_prefetch(A + (size_t)bm * K + (size_t)(kt + 2) * 32 + (size_t)r * K + col, 0, 1);
            __builtin_prefetch(Bt + (size_t)bn * K + (size_t)(kt + 2) * 32 + (size_t)r * K + col, 0, 1);
        }
        u16* aT = sA + buf * TILE;
        u16* bT = sB + buf * TILE;
        v16bf af[2], bfv[4];
#pragma unroll
        for (int i = 0; i < 2; ++i) af[i] = ld_frag_a(aT + (wr * 32 + i * 16) * 32, 32);
#pragma unroll
        for (int j = 0; j < 4; ++j) bfv[j] = ld_frag_b(bT + (wc * 64 + j * 16) * 32, 32);
#pragma unroll
        for (int i = 0; i < 2; ++i)
#pragma unroll
            for (int j = 0; j < 4; ++j) acc[i][j] = wmma_bf16(af[i], bfv[j], acc[i][j]);
        stage_fence();
        __syncthreads();
        buf ^= 1;
    }

    int lane = tid & 31, lm = lane & 15, lh = lane >> 4;
#pragma unroll
    for (int i = 0; i < 2; ++i)
#pragma unroll
        for (int j = 0; j < 4; ++j)
#pragma unroll
            for (int r = 0; r < 8; ++r) {
                int row = bm + wr * 32 + i * 16 + r + 8 * lh;
                int col = bn + wc * 64 + j * 16 + lm;
                C[(size_t)row * N + col] = acc[i][j][r];
            }
}

// ---------------- RoPE + head split: QKV f32 [T][3072] -> Qr/Kr/Vv bf16 ----
__device__ __forceinline__ float rope_val(const float* hd, int d, int t) {
    int i = d & 63;
    // inv_freq = 10000^(-i/64) = 2^(-i*log2(10000)/64)
    float inv = exp2f(-(float)i * (13.287712379549449f / 64.0f));
    float ang = (float)t * inv;
    float sn, cs;
    sincosf(ang, &sn, &cs);
    float x1 = hd[i], x2 = hd[i + 64];
    return (d < 64) ? (x1 * cs - x2 * sn) : (x2 * cs + x1 * sn);
}
__global__ void rope_split_kernel(const float* __restrict__ qkv, u16* __restrict__ Qr,
                                  u16* __restrict__ Kr, u16* __restrict__ Vv) {
    int idx = blockIdx.x * blockDim.x + threadIdx.x;   // T * 3072
    if (idx >= SEQ_T * QKV_N) return;
    int t = idx / QKV_N, c = idx - t * QKV_N;
    const float* row = qkv + (size_t)t * QKV_N;
    if (c < N_HEADS * D_HEAD) {
        int h = c >> 7, d = c & 127;
        Qr[((size_t)h * SEQ_T + t) * D_HEAD + d] = f2bf(rope_val(row + h * D_HEAD, d, t));
    } else if (c < N_HEADS * D_HEAD + N_KV_HEADS * D_HEAD) {
        int cc = c - N_HEADS * D_HEAD;
        int h = cc >> 7, d = cc & 127;
        Kr[((size_t)h * SEQ_T + t) * D_HEAD + d] =
            f2bf(rope_val(row + N_HEADS * D_HEAD + h * D_HEAD, d, t));
    } else {
        int cc = c - (N_HEADS * D_HEAD + N_KV_HEADS * D_HEAD);
        int h = cc >> 7, d = cc & 127;
        Vv[((size_t)h * SEQ_T + t) * D_HEAD + d] = f2bf(row[c]);
    }
}

// ---------------- fused flash attention (one head x one 128-row Q block per WG) ----
// 8 waves, wave = 16 Q rows. Scores, rank-32 bias matrix, and PV all via WMMA.
__global__ __launch_bounds__(256) void flash_attn_kernel(
    const u16* __restrict__ Qr, const u16* __restrict__ Kr, const u16* __restrict__ Vv,
    const u16* __restrict__ Abf, const u16* __restrict__ Bbf,
    const float* __restrict__ pol_dir, const float* __restrict__ pol_gate,
    const float* __restrict__ gtp_gamma, u16* __restrict__ O) {
    const int T = SEQ_T, D = D_HEAD;
    int qb = blockIdx.x;            // 0..15 Q row block
    int h = blockIdx.y;             // 0..15 head
    int kvh = h >> 2;               // GQA group

    extern __shared__ char smem[];
    u16* sK  = (u16*)smem;          // [128][128] K block (t x d)
    u16* sVt = sK + 128 * 128;      // [128][128] V^T (d x t)
    u16* sB  = sVt + 128 * 128;     // [128][32]  bias B rows (also A staging at init)
    u16* sP  = sB + 128 * 32;       // [8][16][128] Q staging then per-wave P staging

    int tid = threadIdx.x, wave = tid >> 5, lane = tid & 31;
    int lm = lane & 15, lh = lane >> 4;

    float pol = fminf(fmaxf(pol_dir[h], -1.f), 1.f);
    float g = gtp_gamma[h];
    float gamma = fmaxf((g > 20.f) ? g : log1pf(expf(g)), 1e-6f);
    float gate = 1.f / (1.f + __expf(-pol_gate[h]));
    const float scale = 0.08838834764831845f;  // 1/sqrt(128)

    // ---- stage Q block and bias-A rows, pull persistent fragments ----
    {
        const u16* Qblk = Qr + ((size_t)h * T + qb * 128) * D;
#pragma unroll
        for (int c = 0; c < 8; ++c) {
            int idx = tid + c * 256;
            int r = idx >> 4, col = (idx & 15) * 8;
            stage16(sP + r * 128 + col, Qblk + (size_t)r * D + col);
        }
        const u16* Ablk = Abf + (size_t)qb * 128 * RANK;
#pragma unroll
        for (int c = 0; c < 2; ++c) {
            int idx = tid + c * 256;
            int r = idx >> 2, col = (idx & 3) * 8;
            stage16(sB + r * 32 + col, Ablk + (size_t)r * RANK + col);
        }
    }
    stage_fence();
    __syncthreads();
    v16bf qf[4];
#pragma unroll
    for (int kc = 0; kc < 4; ++kc) qf[kc] = ld_frag_a(sP + (wave * 16) * 128 + kc * 32, 128);
    v16bf apf = ld_frag_a(sB + (wave * 16) * 32, 32);
    __syncthreads();

    v8f o[8];
    float mrun[8], lrun[8];
#pragma unroll
    for (int db = 0; db < 8; ++db) o[db] = v8f_zero();
#pragma unroll
    for (int r = 0; r < 8; ++r) { mrun[r] = -INFINITY; lrun[r] = 0.f; }

    int irow0 = qb * 128 + wave * 16;

    for (int j = 0; j <= qb; ++j) {
        // ---- stage K (async, row-major), V (b128 load + transpose scatter), bias-B ----
        const u16* Kblk = Kr + ((size_t)kvh * T + j * 128) * D;
        const u16* Vblk = Vv + ((size_t)kvh * T + j * 128) * D;
#pragma unroll
        for (int c = 0; c < 8; ++c) {
            int idx = tid + c * 256;
            int r = idx >> 4, col = (idx & 15) * 8;
            stage16(sK + r * 128 + col, Kblk + (size_t)r * D + col);
            union { uint4 q; u16 e[8]; } vv;
            vv.q = *(const uint4*)(Vblk + (size_t)r * D + col);
#pragma unroll
            for (int e = 0; e < 8; ++e) sVt[(col + e) * 128 + r] = vv.e[e];
        }
        const u16* Bblk = Bbf + (size_t)j * 128 * RANK;
#pragma unroll
        for (int c = 0; c < 2; ++c) {
            int idx = tid + c * 256;
            int r = idx >> 2, col = (idx & 3) * 8;
            stage16(sB + r * 32 + col, Bblk + (size_t)r * RANK + col);
        }
        stage_fence();
        __syncthreads();

        // ---- scores + bias ----
        v8f sfr[8];
#pragma unroll
        for (int nb = 0; nb < 8; ++nb) {
            v8f s = v8f_zero();
#pragma unroll
            for (int kc = 0; kc < 4; ++kc) {
                v16bf kf = ld_frag_b(sK + (nb * 16) * 128 + kc * 32, 128);
                s = wmma_bf16(qf[kc], kf, s);
            }
            v8f mm = v8f_zero();
            {
                v16bf bpf = ld_frag_b(sB + (nb * 16) * 32, 32);
                mm = wmma_bf16(apf, bpf, mm);   // rank-32 bias logits in one WMMA
            }
            int jcol = j * 128 + nb * 16 + lm;
#pragma unroll
            for (int r = 0; r < 8; ++r) {
                int i = irow0 + r + 8 * lh;
                float delta = (float)(i - jcol);
                float sig = 1.f / (1.f + __expf(-mm[r]));
                float bias = -pol * delta * (1.f / 4096.f) - gamma * delta + gate * sig;
                float val = s[r] * scale + bias;
                s[r] = (i >= jcol) ? val : -1.0e30f;
            }
            sfr[nb] = s;
        }

        // ---- online softmax (row stats live in the 16-lane half-group) ----
#pragma unroll
        for (int r = 0; r < 8; ++r) {
            float v = sfr[0][r];
#pragma unroll
            for (int nb = 1; nb < 8; ++nb) v = fmaxf(v, sfr[nb][r]);
#pragma unroll
            for (int off = 8; off >= 1; off >>= 1) v = fmaxf(v, __shfl_xor(v, off, 32));
            float mnew = fmaxf(mrun[r], v);
            float corr = __expf(mrun[r] - mnew);
            mrun[r] = mnew;
            lrun[r] *= corr;
#pragma unroll
            for (int db = 0; db < 8; ++db) o[db][r] *= corr;
            float ls = 0.f;
#pragma unroll
            for (int nb = 0; nb < 8; ++nb) {
                float p = __expf(sfr[nb][r] - mnew);
                sfr[nb][r] = p;
                ls += p;
            }
#pragma unroll
            for (int off = 8; off >= 1; off >>= 1) ls += __shfl_xor(ls, off, 32);
            lrun[r] += ls;
        }

        // ---- P: D-layout -> LDS -> A-layout (wave-private region) ----
        u16* sPw = sP + wave * 16 * 128;
#pragma unroll
        for (int nb = 0; nb < 8; ++nb)
#pragma unroll
            for (int r = 0; r < 8; ++r)
                sPw[(r + 8 * lh) * 128 + nb * 16 + lm] = f2bf(sfr[nb][r]);

        v16bf pf[4];
#pragma unroll
        for (int kc = 0; kc < 4; ++kc) pf[kc] = ld_frag_a(sPw + kc * 32, 128);
#pragma unroll
        for (int db = 0; db < 8; ++db) {
            v8f acc = o[db];
#pragma unroll
            for (int kc = 0; kc < 4; ++kc) {
                v16bf vf = ld_frag_b(sVt + (db * 16) * 128 + kc * 32, 128);
                acc = wmma_bf16(pf[kc], vf, acc);
            }
            o[db] = acc;
        }
        __syncthreads();
    }

    // ---- epilogue: normalize and store bf16 [T][H*D] ----
#pragma unroll
    for (int r = 0; r < 8; ++r) {
        float inv = 1.f / lrun[r];
        int i = irow0 + r + 8 * lh;
#pragma unroll
        for (int db = 0; db < 8; ++db)
            O[(size_t)i * (N_HEADS * D_HEAD) + h * D_HEAD + db * 16 + lm] = f2bf(o[db][r] * inv);
    }
}

// ---------------- host orchestration ----------------
extern "C" void kernel_launch(void* const* d_in, const int* in_sizes, int n_in,
                              void* d_out, int out_size, void* d_ws, size_t ws_size,
                              hipStream_t stream) {
    const float* x        = (const float*)d_in[0];
    const float* Wq       = (const float*)d_in[1];
    const float* Wk       = (const float*)d_in[2];
    const float* Wv       = (const float*)d_in[3];
    const float* Wo       = (const float*)d_in[4];
    const float* pol_dir  = (const float*)d_in[5];
    const float* pol_WA   = (const float*)d_in[6];
    const float* pol_WB   = (const float*)d_in[7];
    const float* pol_gate = (const float*)d_in[8];
    const float* gtp_gamma= (const float*)d_in[9];

    char* ws = (char*)d_ws;
    size_t off = 0;
    auto alloc = [&](size_t bytes) { char* p = ws + off; off = (off + bytes + 255) & ~(size_t)255; return p; };

    u16*   x_bf   = (u16*)  alloc((size_t)SEQ_T * D_MODEL * 2);
    u16*   wqkvt  = (u16*)  alloc((size_t)QKV_N * D_MODEL * 2);     // [3072][2048] bf16 (Wq^T|Wk^T|Wv^T)
    u16*   wot    = (u16*)  alloc((size_t)D_MODEL * D_MODEL * 2);   // Wo^T
    float* qkvf   = (float*)alloc((size_t)SEQ_T * QKV_N * 4);
    u16*   Qrope  = (u16*)  alloc((size_t)N_HEADS * SEQ_T * D_HEAD * 2);
    u16*   Krope  = (u16*)  alloc((size_t)N_KV_HEADS * SEQ_T * D_HEAD * 2);
    u16*   Vval   = (u16*)  alloc((size_t)N_KV_HEADS * SEQ_T * D_HEAD * 2);
    u16*   Abf    = (u16*)  alloc((size_t)SEQ_T * RANK * 2);
    u16*   Bbf    = (u16*)  alloc((size_t)SEQ_T * RANK * 2);
    u16*   attn   = (u16*)  alloc((size_t)SEQ_T * N_HEADS * D_HEAD * 2);

    // 1) conversions
    {
        int n = SEQ_T * D_MODEL;
        cvt_bf16_kernel<<<(n + 255) / 256, 256, 0, stream>>>(x, x_bf, n);
        n = D_MODEL * (N_HEADS * D_HEAD);
        transpose_cvt_kernel<<<(n + 255) / 256, 256, 0, stream>>>(Wq, wqkvt, D_MODEL, N_HEADS * D_HEAD);
        n = D_MODEL * (N_KV_HEADS * D_HEAD);
        transpose_cvt_kernel<<<(n + 255) / 256, 256, 0, stream>>>(
            Wk, wqkvt + (size_t)(N_HEADS * D_HEAD) * D_MODEL, D_MODEL, N_KV_HEADS * D_HEAD);
        transpose_cvt_kernel<<<(n + 255) / 256, 256, 0, stream>>>(
            Wv, wqkvt + (size_t)(N_HEADS * D_HEAD + N_KV_HEADS * D_HEAD) * D_MODEL, D_MODEL,
            N_KV_HEADS * D_HEAD);
        n = D_MODEL * D_MODEL;
        transpose_cvt_kernel<<<(n + 255) / 256, 256, 0, stream>>>(Wo, wot, D_MODEL, D_MODEL);
    }

    // 2) rank-32 bias factors (fp32 accumulate, bf16 out)
    {
        int n = SEQ_T * RANK;
        pol_ab_kernel<<<(n + 255) / 256, 256, 0, stream>>>(x, pol_WA, pol_WB, Abf, Bbf);
    }

    // 3) fused QKV projection: [2048,2048] x [2048,3072]
    {
        dim3 grid(QKV_N / 128, SEQ_T / 128);
        gemm_tn_bf16<<<grid, 256, 32768, stream>>>(x_bf, wqkvt, qkvf, SEQ_T, QKV_N, D_MODEL);
    }

    // 4) RoPE + head split
    {
        int n = SEQ_T * QKV_N;
        rope_split_kernel<<<(n + 255) / 256, 256, 0, stream>>>(qkvf, Qrope, Krope, Vval);
    }

    // 5) fused flash attention (16 q-blocks x 16 heads)
    {
        dim3 grid(SEQ_T / 128, N_HEADS);
        size_t lds = (size_t)(128 * 128 + 128 * 128 + 128 * 32 + 8 * 16 * 128) * 2;  // 104 KB
        flash_attn_kernel<<<grid, 256, lds, stream>>>(Qrope, Krope, Vval, Abf, Bbf, pol_dir,
                                                      pol_gate, gtp_gamma, attn);
    }

    // 6) output projection -> fp32 d_out
    {
        dim3 grid(D_MODEL / 128, SEQ_T / 128);
        gemm_tn_bf16<<<grid, 256, 32768, stream>>>(attn, wot, (float*)d_out, SEQ_T, D_MODEL,
                                                   D_MODEL);
    }
    (void)in_sizes; (void)n_in; (void)out_size; (void)ws_size;
}
